// BertEncoderModel_74629351735814
// MI455X (gfx1250) — matrix operand order repaired and checked by reading, not compiled, not run on hardware
//
#include <hip/hip_runtime.h>
#include <hip/hip_bf16.h>

typedef __attribute__((ext_vector_type(16))) _Float16 v16h;
typedef __attribute__((ext_vector_type(8)))  float    v8f;

#define TB 256
#define BM 128
#define BN 64
#define KT 32
#define LDA_S 40   // padded LDS row stride (halves) -> conflict-free b128 frag loads
#define LDB_S 40

static __device__ inline v16h load_frag(const _Float16* p0, const _Float16* p1) {
    v16h v;
    __builtin_memcpy(&v, p0, 16);
    __builtin_memcpy(((char*)&v) + 16, p1, 16);
    return v;
}

static __device__ inline unsigned int pk2(float x, float y) {
    return __builtin_bit_cast(unsigned int, __builtin_amdgcn_cvt_pkrtz(x, y));
}

// C[M,N] = epilogue( alpha * A[M,K] x B + bias )  (batched, strided)
// BT=1: B is [N,K] row-major;  BT=0: B is [K,N] row-major.
// Requirements: M % 128 == 0, K % 32 == 0, (!NBOUND -> N % 64 == 0),
//               A/B rows 16-byte aligned (lda/ldb % 4 == 0, bases 16B-aligned).
// act: 0=none, 1=exact gelu. rowScale: optional per-row gate, C (+)= scale*val.
template<int BT, bool NBOUND>
__global__ __launch_bounds__(TB)
void gemm_wmma(const float* __restrict__ A, const float* __restrict__ B,
               const float* __restrict__ bias, float* __restrict__ C,
               const float* __restrict__ rowScale, int rsStride,
               int M, int N, int K, int lda, int ldb, int ldc,
               int batchMod,
               long long sA, long long sA2, long long sB, long long sB2,
               long long sC, long long sC2,
               float alpha, int act, int accumulate)
{
    __shared__ _Float16 As[2][BM * LDA_S];
    __shared__ _Float16 Bs[2][BN * LDB_S];

    const int bz  = blockIdx.z;
    const int bzo = bz / batchMod, bzi = bz % batchMod;
    A += bzo * sA + bzi * sA2;
    B += bzo * sB + bzi * sB2;
    C += bzo * sC + bzi * sC2;

    const int m0 = blockIdx.y * BM;
    const int n0 = blockIdx.x * BN;
    const int tid  = threadIdx.x;
    const int lane = tid & 31;
    const int wave = tid >> 5;
    const int wm = (wave >> 1) * 32;   // 4 waves along M
    const int wn = (wave &  1) * 32;   // 2 waves along N

    // staging coordinates (unconditional, vectorized)
    const int aRow = tid >> 1;             // 0..127
    const int aCol = (tid & 1) * 16;       // float offset within K-tile
    // BT=1: per i: n=idx>>3 (0..63), c4=idx&7 -> 4 floats along K
    // BT=0: per i: kk=idx>>4 (0..31), c4=idx&15 -> 4 floats along N

    float4 aR[4];
    float4 bR[2];

    auto loadG = [&](int k0) {
        const float* Arow = A + (long long)(m0 + aRow) * lda + k0 + aCol;
        #pragma unroll
        for (int i = 0; i < 4; ++i)
            aR[i] = *(const float4*)(Arow + i * 4);
        #pragma unroll
        for (int i = 0; i < 2; ++i) {
            int idx = i * TB + tid;
            if (BT) {
                int n = idx >> 3, c4 = idx & 7;
                int gn = n0 + n;
                if (!NBOUND || gn < N)
                    bR[i] = *(const float4*)(B + (long long)gn * ldb + k0 + c4 * 4);
                else
                    bR[i] = make_float4(0.f, 0.f, 0.f, 0.f);
            } else {
                int kk = idx >> 4, c4 = idx & 15;
                bR[i] = *(const float4*)(B + (long long)(k0 + kk) * ldb + n0 + c4 * 4);
            }
        }
    };

    auto stage = [&](int buf) {
        // A: 16 consecutive halves -> two b128 stores
        uint4 p0 = make_uint4(pk2(aR[0].x, aR[0].y), pk2(aR[0].z, aR[0].w),
                              pk2(aR[1].x, aR[1].y), pk2(aR[1].z, aR[1].w));
        uint4 p1 = make_uint4(pk2(aR[2].x, aR[2].y), pk2(aR[2].z, aR[2].w),
                              pk2(aR[3].x, aR[3].y), pk2(aR[3].z, aR[3].w));
        _Float16* ap = &As[buf][aRow * LDA_S + aCol];
        *(uint4*)ap       = p0;
        *(uint4*)(ap + 8) = p1;
        #pragma unroll
        for (int i = 0; i < 2; ++i) {
            int idx = i * TB + tid;
            if (BT) {
                int n = idx >> 3, c4 = idx & 7;
                uint2 q = make_uint2(pk2(bR[i].x, bR[i].y), pk2(bR[i].z, bR[i].w));
                *(uint2*)&Bs[buf][n * LDB_S + c4 * 4] = q;
            } else {
                int kk = idx >> 4, c4 = idx & 15;
                Bs[buf][(c4 * 4 + 0) * LDB_S + kk] = (_Float16)bR[i].x;
                Bs[buf][(c4 * 4 + 1) * LDB_S + kk] = (_Float16)bR[i].y;
                Bs[buf][(c4 * 4 + 2) * LDB_S + kk] = (_Float16)bR[i].z;
                Bs[buf][(c4 * 4 + 3) * LDB_S + kk] = (_Float16)bR[i].w;
            }
        }
    };

    v8f acc[2][2];
    const v8f vzero = {0.f,0.f,0.f,0.f,0.f,0.f,0.f,0.f};
    acc[0][0] = vzero; acc[0][1] = vzero; acc[1][0] = vzero; acc[1][1] = vzero;

    const int arow  = wm + (lane & 15);
    const int abase = (lane >> 4) * 8;      // A frag K-base per ISA 16-bit layout
    const int kbB   = (lane >> 4) * 8;      // B frag K-base per ISA 16-bit layout

    loadG(0);
    stage(0);
    __syncthreads();
    int cur = 0;

    for (int k0 = 0; k0 < K; k0 += KT) {
        const bool hasNext = (k0 + KT) < K;
        if (hasNext) loadG(k0 + KT);   // in flight while we compute

        v16h aF[2], bF[2];
        #pragma unroll
        for (int mi = 0; mi < 2; ++mi) {
            const _Float16* p = &As[cur][(arow + mi * 16) * LDA_S + abase];
            aF[mi] = load_frag(p, p + 16);           // K = base..+7, base+16..+23
        }
        #pragma unroll
        for (int ni = 0; ni < 2; ++ni) {
            const _Float16* p = &Bs[cur][(wn + ni * 16 + (lane & 15)) * LDB_S + kbB];
            bF[ni] = load_frag(p, p + 16);           // K = kb..+7, kb+16..+23
        }
        #pragma unroll
        for (int mi = 0; mi < 2; ++mi)
            #pragma unroll
            for (int ni = 0; ni < 2; ++ni)
                acc[mi][ni] = __builtin_amdgcn_wmma_f32_16x16x32_f16(
                    false, aF[mi], false, bF[ni], (short)0, acc[mi][ni], false, false);

        if (hasNext) {
            stage(cur ^ 1);
            __syncthreads();
            cur ^= 1;
        }
    }

    // C/D layout: elem r -> M = r + (lane>=16 ? 8 : 0), N = lane & 15
    const int rlo = lane & 15;
    const int rhi = (lane >> 4) * 8;
    #pragma unroll
    for (int mi = 0; mi < 2; ++mi)
        #pragma unroll
        for (int ni = 0; ni < 2; ++ni)
            #pragma unroll
            for (int r = 0; r < 8; ++r) {
                int gm = m0 + wm + mi * 16 + rhi + r;
                int gn = n0 + wn + ni * 16 + rlo;
                if (!NBOUND || gn < N) {
                    float v = acc[mi][ni][r] * alpha;
                    if (bias) v += bias[gn];
                    if (act == 1) v = 0.5f * v * (1.0f + erff(v * 0.70710678118f));
                    if (rowScale) v *= rowScale[(long long)gm * rsStride];
                    long long off = (long long)gm * ldc + gn;
                    if (accumulate) C[off] += v; else C[off] = v;
                }
            }
}

// ---- embeddings + LayerNorm ----
__global__ __launch_bounds__(TB)
void embed_ln_kernel(const int* __restrict__ ids, const float* __restrict__ tok,
                     const float* __restrict__ pos, const float* __restrict__ seg,
                     const float* __restrict__ g, const float* __restrict__ b,
                     float* __restrict__ x)
{
    __shared__ float red[TB];
    int t = blockIdx.x, s = t & 255, tid = threadIdx.x;
    int id = ids[t];
    float v[3]; float sum = 0.f;
    #pragma unroll
    for (int i = 0; i < 3; ++i) {
        int h = tid + i * 256;
        v[i] = tok[(long long)id * 768 + h] + pos[s * 768 + h] + seg[h];
        sum += v[i];
    }
    red[tid] = sum; __syncthreads();
    for (int st = 128; st > 0; st >>= 1) { if (tid < st) red[tid] += red[tid + st]; __syncthreads(); }
    float mean = red[0] * (1.0f / 768.0f); __syncthreads();
    float sq = 0.f;
    #pragma unroll
    for (int i = 0; i < 3; ++i) { float d = v[i] - mean; sq += d * d; }
    red[tid] = sq; __syncthreads();
    for (int st = 128; st > 0; st >>= 1) { if (tid < st) red[tid] += red[tid + st]; __syncthreads(); }
    float rstd = rsqrtf(red[0] * (1.0f / 768.0f) + 1e-5f);
    #pragma unroll
    for (int i = 0; i < 3; ++i) {
        int h = tid + i * 256;
        x[(long long)t * 768 + h] = (v[i] - mean) * rstd * g[h] + b[h];
    }
}

// ---- x = LN(res + delta) (in-place safe) ----
__global__ __launch_bounds__(TB)
void add_ln_kernel(float* __restrict__ res, const float* __restrict__ delta,
                   const float* __restrict__ g, const float* __restrict__ b)
{
    __shared__ float red[TB];
    int t = blockIdx.x, tid = threadIdx.x;
    float v[3]; float sum = 0.f;
    #pragma unroll
    for (int i = 0; i < 3; ++i) {
        int h = tid + i * 256;
        v[i] = res[(long long)t * 768 + h] + delta[(long long)t * 768 + h];
        sum += v[i];
    }
    red[tid] = sum; __syncthreads();
    for (int st = 128; st > 0; st >>= 1) { if (tid < st) red[tid] += red[tid + st]; __syncthreads(); }
    float mean = red[0] * (1.0f / 768.0f); __syncthreads();
    float sq = 0.f;
    #pragma unroll
    for (int i = 0; i < 3; ++i) { float d = v[i] - mean; sq += d * d; }
    red[tid] = sq; __syncthreads();
    for (int st = 128; st > 0; st >>= 1) { if (tid < st) red[tid] += red[tid + st]; __syncthreads(); }
    float rstd = rsqrtf(red[0] * (1.0f / 768.0f) + 1e-5f);
    #pragma unroll
    for (int i = 0; i < 3; ++i) {
        int h = tid + i * 256;
        res[(long long)t * 768 + h] = (v[i] - mean) * rstd * g[h] + b[h];
    }
}

// ---- masked softmax over last dim (row = b*NH*S + n*S + q) ----
__global__ __launch_bounds__(TB)
void softmax_mask_kernel(float* __restrict__ sc, const int* __restrict__ ids)
{
    __shared__ float red[TB];
    int row = blockIdx.x, tid = threadIdx.x;
    int b = row / (12 * 256);
    float v = sc[(long long)row * 256 + tid];
    if (ids[b * 256 + tid] == 0) v = -1e9f;
    red[tid] = v; __syncthreads();
    for (int st = 128; st > 0; st >>= 1) { if (tid < st) red[tid] = fmaxf(red[tid], red[tid + st]); __syncthreads(); }
    float mx = red[0]; __syncthreads();
    float e = __expf(v - mx);
    red[tid] = e; __syncthreads();
    for (int st = 128; st > 0; st >>= 1) { if (tid < st) red[tid] += red[tid + st]; __syncthreads(); }
    sc[(long long)row * 256 + tid] = e / red[0];
}

// ---- router: logits, softmax probs -> importance atomics, top-2 gating ----
__global__ __launch_bounds__(TB)
void router_kernel(const float* __restrict__ x, const float* __restrict__ Wr,
                   const float* __restrict__ br, float* __restrict__ gate,
                   float* __restrict__ imp)
{
    __shared__ float red[5][TB];
    int t = blockIdx.x, tid = threadIdx.x;
    float a[5] = {0.f, 0.f, 0.f, 0.f, 0.f};
    #pragma unroll
    for (int i = 0; i < 3; ++i) {
        int h = tid + i * 256;
        float xv = x[(long long)t * 768 + h];
        #pragma unroll
        for (int e = 0; e < 5; ++e) a[e] += xv * Wr[e * 768 + h];
    }
    #pragma unroll
    for (int e = 0; e < 5; ++e) red[e][tid] = a[e];
    __syncthreads();
    for (int st = 128; st > 0; st >>= 1) {
        if (tid < st)
            #pragma unroll
            for (int e = 0; e < 5; ++e) red[e][tid] += red[e][tid + st];
        __syncthreads();
    }
    if (tid == 0) {
        float lg[5], mx = -1e30f;
        #pragma unroll
        for (int e = 0; e < 5; ++e) { lg[e] = red[e][0] + br[e]; mx = fmaxf(mx, lg[e]); }
        float pr[5], se = 0.f;
        #pragma unroll
        for (int e = 0; e < 5; ++e) { pr[e] = __expf(lg[e] - mx); se += pr[e]; }
        #pragma unroll
        for (int e = 0; e < 5; ++e) atomicAdd(&imp[e], pr[e] / se);
        int i1 = 0;
        for (int e = 1; e < 5; ++e) if (lg[e] > lg[i1]) i1 = e;
        int i2 = -1;
        for (int e = 0; e < 5; ++e) { if (e == i1) continue; if (i2 < 0 || lg[e] > lg[i2]) i2 = e; }
        float m2 = fmaxf(lg[i1], lg[i2]);
        float e1 = __expf(lg[i1] - m2), e2 = __expf(lg[i2] - m2);
        float s2 = e1 + e2;
        float gv[5] = {0.f, 0.f, 0.f, 0.f, 0.f};
        gv[i1] = e1 / s2; gv[i2] = e2 / s2;
        #pragma unroll
        for (int e = 0; e < 5; ++e) gate[t * 5 + e] = gv[e];
    }
}

__global__ void init_kernel(float* imp, float* aux) {
    int t = threadIdx.x;
    if (t < 5) imp[t] = 0.f;
    if (t == 5) *aux = 0.f;
}

__global__ void aux_kernel(float* imp, float* aux) {
    if (threadIdx.x == 0) {
        float s = 0.f;
        #pragma unroll
        for (int e = 0; e < 5; ++e) { float v = imp[e] * (1.0f / 1024.0f); s += v * v; imp[e] = 0.f; }
        *aux += 5.0f * s;
    }
}

// ---- masked mean pooling ----
__global__ __launch_bounds__(TB)
void pool_kernel(const float* __restrict__ x, const int* __restrict__ ids,
                 float* __restrict__ pooled)
{
    int b = blockIdx.x, tid = threadIdx.x;
    float cnt = 0.f;
    for (int s = 0; s < 256; ++s) cnt += (ids[b * 256 + s] != 0) ? 1.f : 0.f;
    float inv = 1.0f / fmaxf(cnt, 1e-9f);
    #pragma unroll
    for (int i = 0; i < 3; ++i) {
        int h = tid + i * 256;
        float sm = 0.f;
        for (int s = 0; s < 256; ++s)
            if (ids[b * 256 + s] != 0) sm += x[((long long)b * 256 + s) * 768 + h];
        pooled[b * 768 + h] = sm * inv;
    }
}

// ---- NSP head: tanh(pooled@Wn1^T+bn1)@Wn2^T+bn2 ----
__global__ __launch_bounds__(TB)
void nsp_kernel(const float* __restrict__ pooled, const float* __restrict__ Wn1,
                const float* __restrict__ bn1, const float* __restrict__ Wn2,
                const float* __restrict__ bn2, float* __restrict__ out)
{
    __shared__ float h1[768];
    __shared__ float red[TB];
    int b = blockIdx.x, tid = threadIdx.x;
    #pragma unroll
    for (int i = 0; i < 3; ++i) {
        int j = tid + i * 256;
        float acc = 0.f;
        for (int k = 0; k < 768; ++k) acc += pooled[b * 768 + k] * Wn1[(long long)j * 768 + k];
        h1[j] = tanhf(acc + bn1[j]);
    }
    __syncthreads();
    for (int c = 0; c < 2; ++c) {
        float acc = 0.f;
        #pragma unroll
        for (int i = 0; i < 3; ++i) { int j = tid + i * 256; acc += h1[j] * Wn2[c * 768 + j]; }
        red[tid] = acc; __syncthreads();
        for (int st = 128; st > 0; st >>= 1) { if (tid < st) red[tid] += red[tid + st]; __syncthreads(); }
        if (tid == 0) out[b * 2 + c] = red[0] + bn2[c];
        __syncthreads();
    }
}

extern "C" void kernel_launch(void* const* d_in, const int* in_sizes, int n_in,
                              void* d_out, int out_size, void* d_ws, size_t ws_size,
                              hipStream_t stream) {
    const int* ids        = (const int*)d_in[0];
    const float* tok_emb  = (const float*)d_in[1];
    const float* pos_emb  = (const float*)d_in[2];
    const float* seg_emb  = (const float*)d_in[3];
    const float* emb_ln_g = (const float*)d_in[4];
    const float* emb_ln_b = (const float*)d_in[5];
    const float* Wqkv     = (const float*)d_in[6];
    const float* bqkv     = (const float*)d_in[7];
    const float* Wo       = (const float*)d_in[8];
    const float* bo       = (const float*)d_in[9];
    const float* ln1_g    = (const float*)d_in[10];
    const float* ln1_b    = (const float*)d_in[11];
    const float* ln2_g    = (const float*)d_in[12];
    const float* ln2_b    = (const float*)d_in[13];
    const float* Wr       = (const float*)d_in[14];
    const float* br       = (const float*)d_in[15];
    const float* W1       = (const float*)d_in[16];
    const float* b1       = (const float*)d_in[17];
    const float* W2       = (const float*)d_in[18];
    const float* b2       = (const float*)d_in[19];
    const float* Wn1      = (const float*)d_in[20];
    const float* bn1      = (const float*)d_in[21];
    const float* Wn2      = (const float*)d_in[22];
    const float* bn2      = (const float*)d_in[23];
    const float* mlm_bias = (const float*)d_in[24];

    const int Bn = 4, S = 256, H = 768, NH = 12, Lq = 12, V = 30522, Fd = 3072, Ed = 5;
    const int T = Bn * S;              // 1024 tokens

    float* ws = (float*)d_ws;
    float* x      = ws;                         // 1024*768
    float* qkv    = x + (long long)T * 3 * H;   // 1024*2304
    float* sc     = qkv + (long long)T * 3 * H; // scores 48*256*256 == h 1024*3072
    float* ctx    = sc + (long long)48 * S * S; // 1024*768
    float* tmp    = ctx + (long long)T * H;     // 1024*768 (attn_out / ffn accumulator)
    float* gate   = tmp + (long long)T * H;     // 1024*5
    float* imp    = gate + T * Ed;              // 5 (+pad)
    float* pooled = imp + 8;                    // 4*768

    float* mlm  = (float*)d_out;                          // 1024*30522
    float* nsp  = mlm + (long long)T * V;                 // 8
    float* auxp = nsp + Bn * 2;                           // 1

    auto cdiv = [](int a, int b) { return (a + b - 1) / b; };

    init_kernel<<<1, 32, 0, stream>>>(imp, auxp);
    embed_ln_kernel<<<T, TB, 0, stream>>>(ids, tok_emb, pos_emb, seg_emb, emb_ln_g, emb_ln_b, x);

    const long long sTok = (long long)S * 3 * H;  // per-b stride in qkv
    for (int l = 0; l < Lq; ++l) {
        // QKV: [1024,768] x Wqkv[2304,768]^T
        gemm_wmma<1,false><<<dim3(cdiv(3 * H, BN), cdiv(T, BM), 1), TB, 0, stream>>>(
            x, Wqkv + (long long)l * 3 * H * H, bqkv + l * 3 * H, qkv,
            nullptr, 0, T, 3 * H, H, H, H, 3 * H,
            1, 0, 0, 0, 0, 0, 0, 1.0f, 0, 0);
        // scores = (Q K^T) / 8 : batched over (b, nh)
        gemm_wmma<1,false><<<dim3(cdiv(S, BN), cdiv(S, BM), Bn * NH), TB, 0, stream>>>(
            qkv, qkv + H, nullptr, sc,
            nullptr, 0, S, S, 64, 3 * H, 3 * H, S,
            NH, sTok, 64, sTok, 64, (long long)NH * S * S, (long long)S * S,
            0.125f, 0, 0);
        softmax_mask_kernel<<<Bn * NH * S, TB, 0, stream>>>(sc, ids);
        // ctx = attn @ V : batched, B is [K,N]
        gemm_wmma<0,false><<<dim3(cdiv(64, BN), cdiv(S, BM), Bn * NH), TB, 0, stream>>>(
            sc, qkv + 2 * H, nullptr, ctx,
            nullptr, 0, S, 64, S, S, 3 * H, H,
            NH, (long long)NH * S * S, (long long)S * S, sTok, 64,
            (long long)S * H, 64, 1.0f, 0, 0);
        // attn_out = ctx @ Wo^T + bo
        gemm_wmma<1,false><<<dim3(cdiv(H, BN), cdiv(T, BM), 1), TB, 0, stream>>>(
            ctx, Wo + (long long)l * H * H, bo + l * H, tmp,
            nullptr, 0, T, H, H, H, H, H,
            1, 0, 0, 0, 0, 0, 0, 1.0f, 0, 0);
        add_ln_kernel<<<T, TB, 0, stream>>>(x, tmp, ln1_g + l * H, ln1_b + l * H);
        // MoE router + gating
        router_kernel<<<T, TB, 0, stream>>>(x, Wr + (long long)l * Ed * H, br + l * Ed, gate, imp);
        for (int e = 0; e < Ed; ++e) {
            const long long we = ((long long)l * Ed + e);
            // h = gelu(x @ W1[e]^T + b1[e])   [1024,3072]
            gemm_wmma<1,false><<<dim3(cdiv(Fd, BN), cdiv(T, BM), 1), TB, 0, stream>>>(
                x, W1 + we * Fd * H, b1 + we * Fd, sc,
                nullptr, 0, T, Fd, H, H, H, Fd,
                1, 0, 0, 0, 0, 0, 0, 1.0f, 1, 0);
            // ffn (+)= gate[:,e] * (h @ W2[e]^T + b2[e])
            gemm_wmma<1,false><<<dim3(cdiv(H, BN), cdiv(T, BM), 1), TB, 0, stream>>>(
                sc, W2 + we * H * Fd, b2 + we * H, tmp,
                gate + e, Ed, T, H, Fd, Fd, Fd, H,
                1, 0, 0, 0, 0, 0, 0, 1.0f, 0, (e > 0) ? 1 : 0);
        }
        aux_kernel<<<1, 32, 0, stream>>>(imp, auxp);
        add_ln_kernel<<<T, TB, 0, stream>>>(x, tmp, ln2_g + l * H, ln2_b + l * H);
    }

    pool_kernel<<<Bn, TB, 0, stream>>>(x, ids, pooled);
    nsp_kernel<<<Bn, TB, 0, stream>>>(pooled, Wn1, bn1, Wn2, bn2, nsp);
    // MLM head: [1024,768] x tok_emb[30522,768]^T + mlm_bias (N not multiple of 64)
    gemm_wmma<1,true><<<dim3(cdiv(V, BN), cdiv(T, BM), 1), TB, 0, stream>>>(
        x, tok_emb, mlm_bias, mlm,
        nullptr, 0, T, V, H, H, H, V,
        1, 0, 0, 0, 0, 0, 0, 1.0f, 0, 0);
}
